// NeuroModelV2_35648228557609
// MI455X (gfx1250) — matrix operand order repaired and checked by reading, not compiled, not run on hardware
//
#include <hip/hip_runtime.h>
#include <hip/hip_bf16.h>
#include <math.h>

// ---------------------------------------------------------------------------
// Types for CDNA5 WMMA (wave32, 16x16x32 bf16 -> f32)
// ---------------------------------------------------------------------------
typedef __bf16 bf16_t;
typedef __attribute__((ext_vector_type(16))) __bf16 v16bf;
typedef __attribute__((ext_vector_type(8)))  float  v8f;

union F16x16 { v16bf v; uint4 q[2]; };

__device__ __forceinline__ bf16_t f2bf(float f) {
  unsigned u = __builtin_bit_cast(unsigned, f);
  unsigned r = (u + 0x7FFFu + ((u >> 16) & 1u)) >> 16;  // round-to-nearest-even
  unsigned short s = (unsigned short)r;
  return __builtin_bit_cast(bf16_t, s);
}

__device__ __forceinline__ float geluf(float v) {  // exact GELU (approximate=False)
  return 0.5f * v * (1.0f + erff(v * 0.70710678118654752440f));
}

static constexpr int D     = 1024;
static constexpr int TOK   = 16384;   // B*S = 4*4096
static constexpr int CCLS  = 1000;
static constexpr int SD    = 64;
static constexpr int LNUM  = 6;
static constexpr int KKEEP = 204;     // int(1024 * (1 - 0.8))

// ---------------------------------------------------------------------------
// CDNA5 async copy: memory -> LDS without a VGPR round-trip (ASYNCcnt-tracked)
// VDST operand = 32-bit LDS byte offset (low 32 bits of generic LDS pointer),
// VADDR = 64-bit global address. See CDNA5 ISA / 15.18.3.
// ---------------------------------------------------------------------------
#define ASYNC_B128(gptr, lptr)                                                \
  asm volatile("global_load_async_to_lds_b128 %0, %1, off"                    \
               :: "v"((unsigned)(unsigned long long)(lptr)),                  \
                  "v"((unsigned long long)(gptr))                             \
               : "memory")

template<int N>
__device__ __forceinline__ void wait_asynccnt_le() {
  asm volatile("s_wait_asynccnt %0" :: "i"(N) : "memory");
}

// ---------------------------------------------------------------------------
// WMMA GEMM core: C[128 x BN] tile of A[M,K](bf16,row) * Wt[N,K](bf16,row)^T
// 8 waves: 4 row-groups x 2 col-groups; each wave: 2 row-frags x (BN/32) frags.
// Double-buffered LDS, async DMA for tile k+1 overlapped with WMMA on tile k.
// ---------------------------------------------------------------------------
template<int BN>
__device__ __forceinline__ void gemm_core(const bf16_t* __restrict__ A, int lda,
                                          const bf16_t* __restrict__ Wt, int ldb,
                                          int K, v8f (&acc)[2][BN / 32]) {
  constexpr int CF   = BN / 32;
  constexpr int NOPS = (BN == 128) ? 4 : 3;   // async ops per thread per tile
  __shared__ __align__(16) bf16_t lA[2][128 * 32];
  __shared__ __align__(16) bf16_t lB[2][BN * 32];
  const int t = threadIdx.x;
  const int lane = t & 31, w = t >> 5;
  const int lane16 = lane & 15, laneHi = lane >> 4;
  const int waveRow = w & 3, waveCol = w >> 2;
  const int mBase = blockIdx.y * 128;
  const int nBase = blockIdx.x * BN;

  // Per-thread staging addresses (each thread owns 32B of A and 16/32B of B).
  const int rowA = t >> 1, koA = (t & 1) * 16;
  const bf16_t* gA = A + (size_t)(mBase + rowA) * lda + koA;
  const int rowB = (BN == 128) ? (t >> 1) : (t >> 2);
  const int koB  = (BN == 128) ? ((t & 1) * 16) : ((t & 3) * 8);
  const bf16_t* gB = Wt + (size_t)(nBase + rowB) * ldb + koB;

  auto issue = [&](int buf, int k0) {
    bf16_t* la = &lA[buf][rowA * 32 + koA];
    ASYNC_B128(gA + k0,     la);
    ASYNC_B128(gA + k0 + 8, la + 8);
    bf16_t* lb = &lB[buf][rowB * 32 + koB];
    ASYNC_B128(gB + k0, lb);
    if constexpr (BN == 128) ASYNC_B128(gB + k0 + 8, lb + 8);
  };

  for (int r = 0; r < 2; ++r)
    for (int c = 0; c < CF; ++c)
      for (int e = 0; e < 8; ++e) acc[r][c][e] = 0.f;

  issue(0, 0);
  for (int k0 = 0; k0 < K; k0 += 32) {
    const int buf = (k0 >> 5) & 1;
    if (k0 + 32 < K) {
      issue(buf ^ 1, k0 + 32);               // DMA next tile while computing
      wait_asynccnt_le<NOPS>();              // in-order: current tile landed
    } else {
      wait_asynccnt_le<0>();
    }
    __syncthreads();                         // all waves' portions visible

    // Build fragments per ISA 7.12.2 wave32 layouts.
    F16x16 a[2], b[CF];
    for (int r = 0; r < 2; ++r) {
      // A 16x32: lanes0-15 hold K{0..7,16..23}, lanes16-31 hold K{8..15,24..31}
      const bf16_t* ap = &lA[buf][(waveRow * 32 + r * 16 + lane16) * 32 + laneHi * 8];
      a[r].q[0] = *(const uint4*)(ap);
      a[r].q[1] = *(const uint4*)(ap + 16);
    }
    for (int c = 0; c < CF; ++c) {
      // B 32x16: lanes0-15 hold K{0..7,16..23}, lanes16-31 hold K{8..15,24..31}, N = lane16 (Wt is [N][K])
      const bf16_t* bp = &lB[buf][(waveCol * (BN / 2) + c * 16 + lane16) * 32 + laneHi * 8];
      b[c].q[0] = *(const uint4*)(bp);
      b[c].q[1] = *(const uint4*)(bp + 16);
    }
    for (int r = 0; r < 2; ++r)
      for (int c = 0; c < CF; ++c)
        acc[r][c] = __builtin_amdgcn_wmma_f32_16x16x32_bf16(
            false, a[r].v, false, b[c].v, (short)0, acc[r][c], false, false);
    __syncthreads();                         // done reading buf before reuse
  }
}

// epilogue coordinates (C/D layout: VGPR v -> M = base + laneHi*8 + v, N = base + lane16)
#define GEMM_EPI_COORDS(BN)                                               \
  const int t_ = threadIdx.x;                                             \
  const int lane16_ = t_ & 15, laneHi_ = (t_ >> 4) & 1, w_ = t_ >> 5;     \
  const int mB_ = blockIdx.y * 128 + (w_ & 3) * 32;                       \
  const int nB_ = blockIdx.x * (BN) + (w_ >> 2) * ((BN) / 2);

// ---------------------------------------------------------------------------
// GEMM kernel variants
// ---------------------------------------------------------------------------
__global__ __launch_bounds__(256) void gemm_dca_gelu(
    const bf16_t* __restrict__ A, const bf16_t* __restrict__ Wt,
    const float* __restrict__ bias, float* __restrict__ h) {
  v8f acc[2][4];
  gemm_core<128>(A, D, Wt, D, D, acc);
  GEMM_EPI_COORDS(128);
  for (int r = 0; r < 2; ++r)
    for (int c = 0; c < 4; ++c) {
      const int n = nB_ + c * 16 + lane16_;
      const float bv = bias[n];
      const int m0 = mB_ + r * 16 + laneHi_ * 8;
      for (int v = 0; v < 8; ++v)
        h[(size_t)(m0 + v) * D + n] = geluf(acc[r][c][v] + bv);
    }
}

__global__ __launch_bounds__(256) void gemm_exit_store(
    const bf16_t* __restrict__ A, const bf16_t* __restrict__ Wt,
    const float* __restrict__ bias, float* __restrict__ logits) {
  v8f acc[2][4];
  gemm_core<128>(A, D, Wt, D, D, acc);
  GEMM_EPI_COORDS(128);
  for (int r = 0; r < 2; ++r)
    for (int c = 0; c < 4; ++c) {
      const int n = nB_ + c * 16 + lane16_;
      const float bv = (n < CCLS) ? bias[n] : 0.f;
      const int m0 = mB_ + r * 16 + laneHi_ * 8;
      for (int v = 0; v < 8; ++v)
        logits[(size_t)(m0 + v) * D + n] = acc[r][c][v] + bv;
    }
}

__global__ __launch_bounds__(256) void gemm_cen_score(
    const bf16_t* __restrict__ A, const bf16_t* __restrict__ Wt,
    const float* __restrict__ bias, const float* __restrict__ cohW,
    float* __restrict__ coh) {
  v8f acc[2][4];
  gemm_core<128>(A, D, Wt, D, D, acc);
  GEMM_EPI_COORDS(128);
  for (int r = 0; r < 2; ++r) {
    const int m0 = mB_ + r * 16 + laneHi_ * 8;
    for (int v = 0; v < 8; ++v) {
      float s = 0.f;
      for (int c = 0; c < 4; ++c) {
        const int n = nB_ + c * 16 + lane16_;
        s += geluf(acc[r][c][v] + bias[n]) * cohW[n];
      }
      atomicAdd(&coh[m0 + v], s);
    }
  }
}

__global__ __launch_bounds__(256) void gemm_cen_apply(
    const bf16_t* __restrict__ A, const bf16_t* __restrict__ WtBase,
    const float* __restrict__ biasBase, const int* __restrict__ bestSel,
    float* __restrict__ x, const int* __restrict__ active) {
  const int best = *bestSel;
  const bf16_t* Wt = WtBase + (size_t)best * D * D;
  const float* bias = biasBase + best * D;
  v8f acc[2][4];
  gemm_core<128>(A, D, Wt, D, D, acc);
  GEMM_EPI_COORDS(128);
  for (int r = 0; r < 2; ++r)
    for (int c = 0; c < 4; ++c) {
      const int n = nB_ + c * 16 + lane16_;
      const float bv = bias[n];
      const int m0 = mB_ + r * 16 + laneHi_ * 8;
      for (int v = 0; v < 8; ++v) {
        const int m = m0 + v;
        if (active[m]) x[(size_t)m * D + n] += geluf(acc[r][c][v] + bv);
      }
    }
}

__global__ __launch_bounds__(256) void gemm_enc_relu(
    const bf16_t* __restrict__ A, const bf16_t* __restrict__ Wt,
    const float* __restrict__ bias, bf16_t* __restrict__ z) {
  v8f acc[2][2];
  gemm_core<64>(A, D, Wt, D, D, acc);
  GEMM_EPI_COORDS(64);
  for (int r = 0; r < 2; ++r)
    for (int c = 0; c < 2; ++c) {
      const int n = nB_ + c * 16 + lane16_;
      const float bv = bias[n];
      const int m0 = mB_ + r * 16 + laneHi_ * 8;
      for (int v = 0; v < 8; ++v)
        z[(size_t)(m0 + v) * SD + n] = f2bf(fmaxf(acc[r][c][v] + bv, 0.f));
    }
}

__global__ __launch_bounds__(256) void gemm_dec_sse(
    const bf16_t* __restrict__ Z, const bf16_t* __restrict__ Wt,
    const float* __restrict__ bias, const float* __restrict__ x,
    float* __restrict__ sse) {
  v8f acc[2][4];
  gemm_core<128>(Z, SD, Wt, SD, SD, acc);
  GEMM_EPI_COORDS(128);
  float local = 0.f;
  for (int r = 0; r < 2; ++r)
    for (int c = 0; c < 4; ++c) {
      const int n = nB_ + c * 16 + lane16_;
      const float bv = bias[n];
      const int m0 = mB_ + r * 16 + laneHi_ * 8;
      for (int v = 0; v < 8; ++v) {
        const float d = (acc[r][c][v] + bv) - x[(size_t)(m0 + v) * D + n];
        local += d * d;
      }
    }
  atomicAdd(sse, local);
}

// ---------------------------------------------------------------------------
// Elementwise / per-token kernels
// ---------------------------------------------------------------------------
__global__ void k_convert_x(const float* __restrict__ x, bf16_t* __restrict__ xb, int n) {
  int i = blockIdx.x * blockDim.x + threadIdx.x;
  if (i < n) xb[i] = f2bf(x[i]);
}

// W[K][N] (f32) -> Wt[Npad][K] (bf16), zero-padded columns
__global__ void k_convert_wt(const float* __restrict__ W, bf16_t* __restrict__ Wt,
                             int K, int N, int Npad) {
  int i = blockIdx.x * blockDim.x + threadIdx.x;
  if (i < Npad * K) {
    int n = i / K, k = i - n * K;
    Wt[i] = (n < N) ? f2bf(W[(size_t)k * N + n]) : f2bf(0.f);
  }
}

__global__ void k_init_state(float* depth, int* active, float* sse, float* coh) {
  int i = blockIdx.x * blockDim.x + threadIdx.x;
  if (i < TOK) { depth[i] = 0.f; active[i] = 1; }
  if (i < 3 * TOK) coh[i] = 0.f;
  if (i == 0) *sse = 0.f;
}

// per-token top-K magnitude threshold (binary search) + masked residual add
__global__ __launch_bounds__(256) void k_topk_residual(
    const float* __restrict__ h, float* __restrict__ x, const int* __restrict__ active) {
  __shared__ float sh[D];
  __shared__ float rf[256];
  __shared__ int   ri[256];
  __shared__ float s_lo, s_hi;
  const int tok = blockIdx.x, t = threadIdx.x;
  const float* hr = h + (size_t)tok * D;
  for (int c = t; c < D; c += 256) sh[c] = hr[c];
  __syncthreads();
  float mx = 0.f;
  for (int c = t; c < D; c += 256) mx = fmaxf(mx, fabsf(sh[c]));
  rf[t] = mx; __syncthreads();
  for (int s = 128; s > 0; s >>= 1) { if (t < s) rf[t] = fmaxf(rf[t], rf[t + s]); __syncthreads(); }
  if (t == 0) { s_lo = 0.f; s_hi = rf[0]; }
  __syncthreads();
  for (int it = 0; it < 22; ++it) {
    const float mid = 0.5f * (s_lo + s_hi);
    int c = 0;
    for (int j = t; j < D; j += 256) c += (fabsf(sh[j]) >= mid);
    ri[t] = c; __syncthreads();
    for (int s = 128; s > 0; s >>= 1) { if (t < s) ri[t] += ri[t + s]; __syncthreads(); }
    if (t == 0) { if (ri[0] >= KKEEP) s_lo = mid; else s_hi = mid; }
    __syncthreads();
  }
  if (active[tok]) {
    const float thr = s_lo;
    float* xr = x + (size_t)tok * D;
    for (int c = t; c < D; c += 256) {
      const float hv = sh[c];
      if (fabsf(hv) >= thr) xr[c] += hv;
    }
  }
}

__global__ __launch_bounds__(256) void k_cen_reduce(
    const float* __restrict__ coh, const int* __restrict__ active,
    const float* __restrict__ coh_b, int* __restrict__ bestOut) {
  __shared__ float s0[256], s1[256], s2[256];
  __shared__ int sc[256];
  const int t = threadIdx.x;
  float a0 = 0, a1 = 0, a2 = 0; int cnt = 0;
  for (int tok = t; tok < TOK; tok += 256)
    if (active[tok]) {
      ++cnt;
      a0 += coh[tok]; a1 += coh[TOK + tok]; a2 += coh[2 * TOK + tok];
    }
  s0[t] = a0; s1[t] = a1; s2[t] = a2; sc[t] = cnt; __syncthreads();
  for (int s = 128; s > 0; s >>= 1) {
    if (t < s) { s0[t] += s0[t+s]; s1[t] += s1[t+s]; s2[t] += s2[t+s]; sc[t] += sc[t+s]; }
    __syncthreads();
  }
  if (t == 0) {
    const float cb = coh_b[0], cnts = (float)sc[0];
    const float denom = fmaxf(cnts, 1.0f);
    const float v0 = (s0[0] + cb * cnts) / denom;
    const float v1 = (s1[0] + cb * cnts) / denom;
    const float v2 = (s2[0] + cb * cnts) / denom;
    int best = 0; float bv = v0;
    if (v1 > bv) { bv = v1; best = 1; }
    if (v2 > bv) { bv = v2; best = 2; }
    *bestOut = best;
  }
}

__global__ __launch_bounds__(256) void k_exit_update(
    const float* __restrict__ logits, float* __restrict__ outLogits,
    float* __restrict__ depth, int* __restrict__ active) {
  __shared__ float rf[256];
  __shared__ float s_max, s_sum;
  const int tok = blockIdx.x, t = threadIdx.x;
  const float* lr = logits + (size_t)tok * D;
  float mx = -3.4e38f;
  for (int c = t; c < CCLS; c += 256) mx = fmaxf(mx, lr[c]);
  rf[t] = mx; __syncthreads();
  for (int s = 128; s > 0; s >>= 1) { if (t < s) rf[t] = fmaxf(rf[t], rf[t+s]); __syncthreads(); }
  if (t == 0) s_max = rf[0];
  __syncthreads();
  float sum = 0.f;
  for (int c = t; c < CCLS; c += 256) sum += expf(lr[c] - s_max);
  rf[t] = sum; __syncthreads();
  for (int s = 128; s > 0; s >>= 1) { if (t < s) rf[t] += rf[t+s]; __syncthreads(); }
  if (t == 0) s_sum = rf[0];
  __syncthreads();
  if (active[tok]) {
    float* orow = outLogits + (size_t)tok * CCLS;
    for (int c = t; c < CCLS; c += 256) orow[c] = lr[c];
    if (t == 0) {
      depth[tok] += 1.0f;
      const float conf = 1.0f / s_sum;  // softmax max
      if (!(conf < 0.85f)) active[tok] = 0;
    }
  }
}

__global__ __launch_bounds__(256) void k_finalize(
    const float* __restrict__ depth, const float* __restrict__ sse,
    float* __restrict__ out) {
  __shared__ float rf[256];
  const int t = threadIdx.x;
  float s = 0.f;
  for (int i = t; i < TOK; i += 256) s += depth[i];
  rf[t] = s; __syncthreads();
  for (int k = 128; k > 0; k >>= 1) { if (t < k) rf[t] += rf[t+k]; __syncthreads(); }
  if (t == 0) {
    out[(size_t)TOK * CCLS]     = rf[0] / (float)TOK;
    out[(size_t)TOK * CCLS + 1] = sse[0] / ((float)TOK * (float)D * (float)LNUM);
  }
}

// ---------------------------------------------------------------------------
// Host-side orchestration
// ---------------------------------------------------------------------------
extern "C" void kernel_launch(void* const* d_in, const int* in_sizes, int n_in,
                              void* d_out, int out_size, void* d_ws, size_t ws_size,
                              hipStream_t stream) {
  (void)in_sizes; (void)n_in; (void)out_size; (void)ws_size;
  const float* in_x     = (const float*)d_in[0];
  const float* in_dcaW  = (const float*)d_in[1];
  const float* in_dcaB  = (const float*)d_in[2];
  const float* in_exitW = (const float*)d_in[3];
  const float* in_exitB = (const float*)d_in[4];
  const float* in_cenW  = (const float*)d_in[5];
  const float* in_cenB  = (const float*)d_in[6];
  const float* in_cohW  = (const float*)d_in[7];
  const float* in_cohB  = (const float*)d_in[8];
  const float* in_encW  = (const float*)d_in[9];
  const float* in_encB  = (const float*)d_in[10];
  const float* in_decW  = (const float*)d_in[11];
  const float* in_decB  = (const float*)d_in[12];
  float* out = (float*)d_out;

  char* ws = (char*)d_ws;
  size_t off = 0;
  auto take = [&](size_t b) { size_t r = off; off += (b + 255) & ~(size_t)255; return r; };
  float*  x_cur  = (float*) (ws + take((size_t)TOK * D * 4));
  bf16_t* x_bf   = (bf16_t*)(ws + take((size_t)TOK * D * 2));
  float*  hbuf   = (float*) (ws + take((size_t)TOK * D * 4));
  float*  logits = (float*) (ws + take((size_t)TOK * D * 4));
  bf16_t* z_bf   = (bf16_t*)(ws + take((size_t)TOK * SD * 2));
  bf16_t* dcaWt  = (bf16_t*)(ws + take((size_t)LNUM * D * D * 2));
  bf16_t* exitWt = (bf16_t*)(ws + take((size_t)LNUM * D * D * 2));
  bf16_t* cenWt  = (bf16_t*)(ws + take((size_t)3 * D * D * 2));
  bf16_t* encWt  = (bf16_t*)(ws + take((size_t)SD * D * 2));
  bf16_t* decWt  = (bf16_t*)(ws + take((size_t)D * SD * 2));
  float*  coh    = (float*) (ws + take((size_t)3 * TOK * 4));
  float*  depth  = (float*) (ws + take((size_t)TOK * 4));
  int*    active = (int*)   (ws + take((size_t)TOK * 4));
  float*  sse    = (float*) (ws + take(4));
  int*    best   = (int*)   (ws + take(4));

  const int NX = TOK * D;
  hipMemcpyAsync(x_cur, in_x, (size_t)NX * 4, hipMemcpyDeviceToDevice, stream);
  k_init_state<<<(3 * TOK + 255) / 256, 256, 0, stream>>>(depth, active, sse, coh);

  // Weight conversion to transposed bf16 [N][K]
  const int wtElems = D * D;
  for (int l = 0; l < LNUM; ++l) {
    k_convert_wt<<<(wtElems + 255) / 256, 256, 0, stream>>>(
        in_dcaW + (size_t)l * D * D, dcaWt + (size_t)l * wtElems, D, D, D);
    k_convert_wt<<<(wtElems + 255) / 256, 256, 0, stream>>>(
        in_exitW + (size_t)l * D * CCLS, exitWt + (size_t)l * wtElems, D, CCLS, D);
  }
  for (int n = 0; n < 3; ++n)
    k_convert_wt<<<(wtElems + 255) / 256, 256, 0, stream>>>(
        in_cenW + (size_t)n * D * D, cenWt + (size_t)n * wtElems, D, D, D);
  k_convert_wt<<<(SD * D + 255) / 256, 256, 0, stream>>>(in_encW, encWt, D, SD, SD);
  k_convert_wt<<<(D * SD + 255) / 256, 256, 0, stream>>>(in_decW, decWt, SD, D, D);

  const dim3 gBig(D / 128, TOK / 128);   // (8, 128)
  const dim3 gEnc(SD / 64, TOK / 128);   // (1, 128)
  const int cvtBlocks = (NX + 255) / 256;

  k_convert_x<<<cvtBlocks, 256, 0, stream>>>(x_cur, x_bf, NX);

  for (int i = 0; i < LNUM; ++i) {
    // SparseDCA: h = gelu(x @ W + b); top-K mask; masked residual
    gemm_dca_gelu<<<gBig, 256, 0, stream>>>(
        x_bf, dcaWt + (size_t)i * wtElems, in_dcaB + (size_t)i * D, hbuf);
    k_topk_residual<<<TOK, 256, 0, stream>>>(hbuf, x_cur, active);
    k_convert_x<<<cvtBlocks, 256, 0, stream>>>(x_cur, x_bf, NX);

    if (i == LNUM / 2) {
      // CEN: 3 branch coherence scores (fused, sims never stored)
      for (int n = 0; n < 3; ++n)
        gemm_cen_score<<<gBig, 256, 0, stream>>>(
            x_bf, cenWt + (size_t)n * wtElems, in_cenB + (size_t)n * D,
            in_cohW, coh + (size_t)n * TOK);
      k_cen_reduce<<<1, 256, 0, stream>>>(coh, active, in_cohB, best);
      gemm_cen_apply<<<gBig, 256, 0, stream>>>(
          x_bf, cenWt, in_cenB, best, x_cur, active);
      k_convert_x<<<cvtBlocks, 256, 0, stream>>>(x_cur, x_bf, NX);
    }

    // VLM vicarious student: z = relu(x @ enc + b); sse += ||z @ dec + b - x||^2
    gemm_enc_relu<<<gEnc, 256, 0, stream>>>(x_bf, encWt, in_encB, z_bf);
    gemm_dec_sse<<<gBig, 256, 0, stream>>>(z_bf, decWt, in_decB, x_cur, sse);

    // PMT early-exit head
    gemm_exit_store<<<gBig, 256, 0, stream>>>(
        x_bf, exitWt + (size_t)i * wtElems, in_exitB + (size_t)i * CCLS, logits);
    k_exit_update<<<TOK, 256, 0, stream>>>(logits, out, depth, active);
  }

  k_finalize<<<1, 256, 0, stream>>>(depth, sse, out);
}
